// UR_Model_24283745091978
// MI455X (gfx1250) — matrix operand order repaired and checked by reading, not compiled, not run on hardware
//
#include <hip/hip_runtime.h>
#include <hip/hip_bf16.h>

typedef __attribute__((ext_vector_type(16))) __bf16 v16bf;
typedef __attribute__((ext_vector_type(8)))  float  v8f;

#define B_   128
#define T_   784
#define H_   512
#define G_   2048   // 4*H
#define NH1  256
#define NC_  10

// workspace layout (bytes)
#define WS_WHB   0ull                       // Wh bf16, B-layout:   2,097,152
#define WS_W1B   (2097152ull)               // W1 bf16, B-layout: 205,520,896
#define WS_HSB   (WS_W1B + 205520896ull)    // hs bf16, A-layout: 102,760,448
#define WS_PART  (WS_HSB + 102760448ull)    // fp32 partials:       8,388,608
#define WS_HID   (WS_PART + 8388608ull)     // hidden fp32:           131,072

__device__ __forceinline__ unsigned short f2bf(float f) {
  unsigned u = __float_as_uint(f);
  unsigned r = (u + 0x7FFFu + ((u >> 16) & 1u)) >> 16;   // round-to-nearest-even
  return (unsigned short)r;
}
__device__ __forceinline__ float sigmoidf_(float x) {
  // v_exp + v_rcp, branch-free: exp(-x)->inf => rcp->0 ; exp(-x)->0 => 1
  return __builtin_amdgcn_rcpf(1.0f + __expf(-x));
}
__device__ __forceinline__ float tanhf_(float x) {
  // tanh(x) = 2*sigmoid(2x) - 1, branch-free
  return __builtin_fmaf(2.0f, sigmoidf_(2.0f * x), -1.0f);
}

// ---------------------------------------------------------------------------
// Convert Wh [512 x 2048] fp32 -> bf16 in WMMA B-operand tile layout.
// B element (k, n): tile = (n/16)*16 + (k/32); lane = n%16 + 16*((k%32)/16);
// elem = k%16.  Blob per tile = 32 lanes x 16 bf16 (contiguous per lane).
// ---------------------------------------------------------------------------
__global__ void prep_whb(const float* __restrict__ Wh,
                         unsigned short* __restrict__ WhB) {
  int idx = blockIdx.x * blockDim.x + threadIdx.x;   // 0..1048575
  int k = idx >> 11;          // row (H)
  int n = idx & 2047;         // col (4H)
  int nt = n >> 4, kb = k >> 5;
  int lane = (n & 15) + (((k >> 4) & 1) << 4);
  int e = k & 15;
  WhB[(((nt << 4) + kb) << 9) + (lane << 4) + e] = f2bf(Wh[idx]);
}

// Same for W1 [401408 x 256] (K-tiles: 12544 blocks of 32).
__global__ void prep_w1b(const float* __restrict__ W1,
                         unsigned short* __restrict__ W1B) {
  long long idx = (long long)blockIdx.x * blockDim.x + threadIdx.x;
  int n = (int)(idx & 255);
  long long k = idx >> 8;
  int nt = n >> 4;
  long long kb = k >> 5;
  int lane = (n & 15) + ((((int)k >> 4) & 1) << 4);
  int e = (int)(k & 15);
  W1B[(((long long)nt * 12544 + kb) << 9) + (lane << 4) + e] = f2bf(W1[idx]);
}

// ---------------------------------------------------------------------------
// Recurrent UR-LSTM. 8 workgroups x 1024 threads (32 waves); WG bg owns batch
// rows 16*bg..16*bg+15. h kept in LDS as bf16 A-operand tiles, double
// buffered. Wave w owns base tile bt=w and computes the f/r/u/o gate tiles
// (bt, bt+32, bt+64, bt+96) so gate math is register-local. 8 waves/SIMD
// give deep wave-level hiding of the L2-resident Wh-tile load latency.
// ---------------------------------------------------------------------------
__global__ void __launch_bounds__(1024) lstm_kernel(
    const float* __restrict__ x,            // [128,784] (D=1)
    const float* __restrict__ Wx,           // [2048]
    const float* __restrict__ bias,         // [2048]
    const unsigned short* __restrict__ WhB, // B-layout bf16
    unsigned short* __restrict__ hsB)       // out: A-layout bf16 per (bg,t)
{
  extern __shared__ char smem[];
  unsigned short* hA = (unsigned short*)smem;        // 2 buffers x 8192 bf16
  float* xs = (float*)(smem + 2 * 8192 * 2);         // [784][16] transposed x

  const int tid  = threadIdx.x;
  const int bt   = tid >> 5;      // wave id == base tile (0..31)
  const int lane = tid & 31;
  const int col  = lane & 15;     // N (and C/D column)
  const int rh   = lane >> 4;     // row-half: C/D rows m = rh*8 + r
  const int bg   = blockIdx.x;

  // Preload x tile transposed: xs[t*16+m] = x[16*bg+m, t]
  for (int i = tid; i < 16 * T_; i += 1024) {
    int m = i & 15, t = i >> 4;
    xs[t * 16 + m] = x[(bg * 16 + m) * T_ + t];
  }
  for (int i = tid; i < 8192; i += 1024) hA[i] = 0;  // h_{-1} = 0

  // Per-thread Wx/b constants for this wave's 4 gate columns.
  float wxv[4], bv[4];
  #pragma unroll
  for (int g = 0; g < 4; ++g) {
    int n = (bt + g * 32) * 16 + col;
    wxv[g] = Wx[n];
    bv[g]  = bias[n];
  }

  float creg[8];                                      // cell state in VGPRs
  #pragma unroll
  for (int r = 0; r < 8; ++r) creg[r] = 0.0f;

  // Scatter constants for the A-operand swizzle of h.
  const int kbD = bt >> 1;
  const int eD  = col;
  const int laneAdd = (bt & 1) << 4;                  // +16 if odd tile (k%32>=16)

  __syncthreads();

  // Uniform byte-offset made opaque each iteration so the compiler cannot
  // hoist the (t-invariant) WhB tile loads out of the sequence loop and
  // spill them to scratch -- B tiles must STREAM from L2 every step.
  unsigned wofs = 0;

  for (int t = 0; t < T_; ++t) {
    asm volatile("" : "+s"(wofs));
    const unsigned short* whb = WhB + wofs;

    const int cur = t & 1, nxt = cur ^ 1;
    const unsigned short* hcur = hA + cur * 8192;
    unsigned short* hnxt = hA + nxt * 8192;

    // x values for my C/D rows (m = rh*8 .. rh*8+7)
    float4 xlo = *(const float4*)(xs + t * 16 + rh * 8);
    float4 xhi = *(const float4*)(xs + t * 16 + rh * 8 + 4);
    float xv[8] = {xlo.x, xlo.y, xlo.z, xlo.w, xhi.x, xhi.y, xhi.z, xhi.w};

    v8f accF, accR, accU, accO;
    #pragma unroll
    for (int r = 0; r < 8; ++r) {        // fold xz_t = x*Wx + b into C
      accF[r] = xv[r] * wxv[0] + bv[0];
      accR[r] = xv[r] * wxv[1] + bv[1];
      accU[r] = xv[r] * wxv[2] + bv[2];
      accO[r] = xv[r] * wxv[3] + bv[3];
    }

    for (int kb = 0; kb < 16; ++kb) {    // K = 512 in steps of 32
      v16bf a = *(const v16bf*)(hcur + kb * 512 + lane * 16);
      const unsigned short* bb = whb + (size_t)kb * 512 + (size_t)lane * 16;
      v16bf b0 = *(const v16bf*)(bb + (size_t)(bt     ) * 8192);
      v16bf b1 = *(const v16bf*)(bb + (size_t)(bt + 32) * 8192);
      v16bf b2 = *(const v16bf*)(bb + (size_t)(bt + 64) * 8192);
      v16bf b3 = *(const v16bf*)(bb + (size_t)(bt + 96) * 8192);
      accF = __builtin_amdgcn_wmma_f32_16x16x32_bf16(false, a, false, b0, (short)0, accF, false, false);
      accR = __builtin_amdgcn_wmma_f32_16x16x32_bf16(false, a, false, b1, (short)0, accR, false, false);
      accU = __builtin_amdgcn_wmma_f32_16x16x32_bf16(false, a, false, b2, (short)0, accU, false, false);
      accO = __builtin_amdgcn_wmma_f32_16x16x32_bf16(false, a, false, b3, (short)0, accO, false, false);
    }

    // UR-LSTM gates, c/h update; scatter h into A-operand swizzle.
    #pragma unroll
    for (int r = 0; r < 8; ++r) {
      float f  = sigmoidf_(accF[r]);
      float rr = sigmoidf_(accR[r]);
      float omf = 1.0f - f;
      float g = rr * (1.0f - omf * omf) + (1.0f - rr) * f * f;
      float c = g * creg[r] + (1.0f - g) * tanhf_(accU[r]);
      creg[r] = c;
      float h = sigmoidf_(accO[r]) * tanhf_(c);
      int m = r + rh * 8;
      hnxt[kbD * 512 + (m + laneAdd) * 16 + eD] = f2bf(h);
    }

    __syncthreads();   // h_t fully in hA[nxt]; safe to read next step
    // Coalesced dump of h_t (already A-swizzled) for the MLP GEMM:
    // 1024 threads x 16B = 16KB, exactly one uint4 per thread.
    const uint4* src = (const uint4*)(hA + nxt * 8192);
    uint4* dst = (uint4*)(hsB + ((size_t)(bg * T_ + t) << 13));
    dst[tid] = src[tid];
  }
}

// ---------------------------------------------------------------------------
// MLP1: [128, 401408] @ [401408, 256], K split into 64 chunks of 196 k-steps.
// One wave per (m-tile, n-tile, k-chunk); fp32 partials, reduced later.
// Latency hidden by 8192 single-wave blocks.
// ---------------------------------------------------------------------------
__global__ void __launch_bounds__(32) mlp1_kernel(
    const unsigned short* __restrict__ hsB,
    const unsigned short* __restrict__ W1B,
    float* __restrict__ partials)
{
  const int lane = threadIdx.x;
  const int mt = blockIdx.x;     // 0..7
  const int nt = blockIdx.y;     // 0..15
  const int kc = blockIdx.z;     // 0..63
  const int KB = 12544;          // total k-blocks of 32
  const int kb0 = kc * 196;

  v8f acc = {};
  const unsigned short* aBase = hsB + (size_t)mt * KB * 512 + (size_t)lane * 16;
  const unsigned short* bBase = W1B + (size_t)nt * KB * 512 + (size_t)lane * 16;
  for (int kb = kb0; kb < kb0 + 196; ++kb) {
    v16bf a = *(const v16bf*)(aBase + (size_t)kb * 512);
    v16bf b = *(const v16bf*)(bBase + (size_t)kb * 512);
    acc = __builtin_amdgcn_wmma_f32_16x16x32_bf16(false, a, false, b, (short)0, acc, false, false);
  }
  const int colL = lane & 15, rh = lane >> 4;
  #pragma unroll
  for (int r = 0; r < 8; ++r) {
    int bb = mt * 16 + r + rh * 8;
    int n  = nt * 16 + colL;
    partials[(size_t)kc * (B_ * NH1) + (size_t)bb * NH1 + n] = acc[r];
  }
}

// Fixed-order K-chunk reduction + bias + ReLU (deterministic).
__global__ void reduce_relu(const float* __restrict__ partials,
                            const float* __restrict__ b1,
                            float* __restrict__ hidden)
{
  int idx = blockIdx.x * blockDim.x + threadIdx.x;   // 0..32767
  float s = b1[idx & 255];
  for (int kc = 0; kc < 64; ++kc) s += partials[(size_t)kc * 32768 + idx];
  hidden[idx] = fmaxf(s, 0.0f);
}

// MLP2: [128,256] @ [256,10] + b2 (tiny).
__global__ void mlp2_kernel(const float* __restrict__ hidden,
                            const float* __restrict__ W2,
                            const float* __restrict__ b2,
                            float* __restrict__ out)
{
  int idx = blockIdx.x * blockDim.x + threadIdx.x;
  if (idx >= B_ * NC_) return;
  int bb = idx / NC_, c = idx % NC_;
  float s = b2[c];
  for (int n = 0; n < NH1; ++n) s += hidden[bb * NH1 + n] * W2[n * NC_ + c];
  out[idx] = s;
}

extern "C" void kernel_launch(void* const* d_in, const int* in_sizes, int n_in,
                              void* d_out, int out_size, void* d_ws, size_t ws_size,
                              hipStream_t stream) {
  const float* x  = (const float*)d_in[0];
  const float* Wx = (const float*)d_in[1];
  const float* Wh = (const float*)d_in[2];
  const float* b  = (const float*)d_in[3];
  const float* W1 = (const float*)d_in[4];
  const float* b1 = (const float*)d_in[5];
  const float* W2 = (const float*)d_in[6];
  const float* b2 = (const float*)d_in[7];

  char* ws = (char*)d_ws;
  unsigned short* WhB = (unsigned short*)(ws + WS_WHB);
  unsigned short* W1B = (unsigned short*)(ws + WS_W1B);
  unsigned short* hsB = (unsigned short*)(ws + WS_HSB);
  float* partials     = (float*)(ws + WS_PART);
  float* hidden       = (float*)(ws + WS_HID);

  prep_whb<<<4096, 256, 0, stream>>>(Wh, WhB);
  prep_w1b<<<401408, 256, 0, stream>>>(W1, W1B);

  size_t smem = 2 * 8192 * 2 + (size_t)16 * T_ * 4;   // 32KB hA + 49KB x = ~83KB
  lstm_kernel<<<8, 1024, smem, stream>>>(x, Wx, b, WhB, hsB);

  mlp1_kernel<<<dim3(8, 16, 64), 32, 0, stream>>>(hsB, W1B, partials);
  reduce_relu<<<128, 256, 0, stream>>>(partials, b1, hidden);
  mlp2_kernel<<<5, 256, 0, stream>>>(hidden, W2, b2, (float*)d_out);
}
